// TemporalPatchAttentionV9_22290880266619
// MI455X (gfx1250) — matrix operand order repaired and checked by reading, not compiled, not run on hardware
//
#include <hip/hip_runtime.h>

// CDNA5 / gfx1250 patch-attention.
// One workgroup (256 threads = 8 wave32 waves) per patch.
// TDM tensor_load_to_lds gathers the 8x8x128 patch tiles of q/k/v into LDS
// (f32 staging), waves convert to f16 WMMA layouts, then
// v_wmma_f32_16x16x32_f16 for QK^T and PV, f32 softmax in LDS,
// overlap-add via non-returning global_atomic_add_f32.

typedef _Float16     v16h  __attribute__((ext_vector_type(16)));
typedef _Float16     v8h   __attribute__((ext_vector_type(8)));
typedef float        v8f   __attribute__((ext_vector_type(8)));
typedef unsigned int u32x4 __attribute__((ext_vector_type(4)));
typedef int          i32x4 __attribute__((ext_vector_type(4)));
typedef int          i32x8 __attribute__((ext_vector_type(8)));

#if defined(__has_builtin)
#if __has_builtin(__builtin_amdgcn_tensor_load_to_lds)
#define HAVE_TDM 1
#endif
#endif
#ifndef HAVE_TDM
#define HAVE_TDM 0
#endif

#define PK    8     // patch size
#define PSTR  4     // patch stride
#define NH    31
#define NW    31
#define BATCH 8
#define CH    128
#define IH    128
#define IW    128
#define T     64    // tokens per patch (8x8)
#define LDQ   136   // f16 row stride for qs/ks
#define LDV   72    // f16 row stride for vs
#define LDP   72    // f16 row stride for ps
#define LDS_SM 68   // f32 row stride for score scratch

// dynamic-LDS carve (bytes); all 16B aligned
#define OFF_STQ  0                      // f32 [128][64] staging for q  (32768)
#define OFF_STK  32768                  // f32 [128][64] staging for k  (32768)
#define OFF_STV  65536                  // f32 [128][64] staging for v  (32768)
#define OFF_QS   98304                  // f16 [64][136] q tile / f32 [64][68] scores (17408)
#define OFF_KS   115712                 // f16 [64][136] k tile (17408)
#define OFF_VS   133120                 // f16 [128][72] v^T tile (18432)
#define OFF_PS   151552                 // f16 [64][72] probs (9216)
#define SMEM_BYTES 160768

#if HAVE_TDM
// Issue one TDM descriptor: 3-D tile, X = 8 contiguous f32 (w), Y = 8 rows
// (stride IW), Z = 128 channels (stride IH*IW), into LDS at lds_off.
__device__ inline void tdm_load_tile(const float* gptr, unsigned int lds_off) {
    const unsigned long long ga = (unsigned long long)(uintptr_t)gptr;

    u32x4 g0;
    g0[0] = 1u;                                  // count=1, user-mode, no gather
    g0[1] = lds_off;                             // lds_addr[31:0]
    g0[2] = (unsigned int)ga;                    // global_addr[31:0]
    g0[3] = (unsigned int)((ga >> 32) & 0x1FFFFFFu) | (2u << 30); // addr[56:32], type=2

    i32x8 g1;
    g1[0] = (int)(2u << 16);                     // data_size=2 (4B); no mask/flags
    g1[1] = (int)((unsigned)IW << 16);           // tensor_dim0[15:0] @bits[63:48]
    g1[2] = (int)((unsigned)IH << 16);           // dim0[31:16]=0, tensor_dim1[15:0] @[95:80]
    g1[3] = (int)((unsigned)PK << 16);           // dim1[31:16]=0, tile_dim0=8 @[127:112]
    g1[4] = (int)((unsigned)PK | ((unsigned)CH << 16)); // tile_dim1=8, tile_dim2=128
    g1[5] = (int)(IW);                           // tensor_dim0_stride[31:0] = 128
    g1[6] = (int)(((unsigned)(IH * IW) & 0xFFFFu) << 16); // stride0[47:32]=0, stride1[15:0]
    g1[7] = (int)((unsigned)(IH * IW) >> 16);    // tensor_dim1_stride[47:16]

    i32x4 g2;
    g2[0] = CH;                                  // tensor_dim2 = 128
    g2[1] = 0;                                   // tensor_dim3 (unused)
    g2[2] = IH * IW;                             // tensor_dim2_stride[31:0] = 16384
    g2[3] = 0;                                   // stride2[47:32]=0, tile_dim3=0

    i32x4 g3 = {0, 0, 0, 0};                     // dims 3/4 unused

    i32x8 gx = {0, 0, 0, 0, 0, 0, 0, 0};         // extra group (clang-23 6-arg form)

    __builtin_amdgcn_tensor_load_to_lds(g0, g1, g2, g3, gx, 0);
}
#endif

// --- WMMA fragment loaders (wave32, 16-bit A 16x32 / B 32x16 layouts) ---

// A fragment: lane L(0..15): row m0+L, K halves {k0..k0+7, k0+16..k0+23};
//             lane L+16:     row m0+L, K halves {k0+8..k0+15, k0+24..k0+31}.
__device__ inline v16h load_frag_a(const _Float16* base, int m0, int k0, int ld, int lane) {
    const int hi = (lane & 16) ? 8 : 0;
    const _Float16* p = base + (m0 + (lane & 15)) * ld + k0 + hi;
    union { v16h v; v8h h[2]; } u;
    u.h[0] = *(const v8h*)(p);
    u.h[1] = *(const v8h*)(p + 16);
    return u.v;
}

// B fragment (from row-major B^T tile): lane L(0..15): col n0+L, K {k0..k0+7, k0+16..k0+23};
// lane L+16: col n0+L, K {k0+8..k0+15, k0+24..k0+31}.
__device__ inline v16h load_frag_b(const _Float16* base, int n0, int k0, int ld, int lane) {
    const int hi = (lane & 16) ? 8 : 0;
    const _Float16* p = base + (n0 + (lane & 15)) * ld + k0 + hi;
    union { v16h v; v8h h[2]; } u;
    u.h[0] = *(const v8h*)(p);
    u.h[1] = *(const v8h*)(p + 16);
    return u.v;
}

// C/D accumulator: VGPR i -> row (i + (lane>=16 ? 8:0)), col = lane&15.
__device__ inline void store_acc(float* sm, int tile, v8f acc, int lane) {
    const int mi = tile >> 2, ni = tile & 3;
    const int r0  = mi * 16 + ((lane & 16) ? 8 : 0);
    const int col = ni * 16 + (lane & 15);
#pragma unroll
    for (int i = 0; i < 8; ++i) sm[(r0 + i) * LDS_SM + col] = acc[i];
}

__global__ void __launch_bounds__(256)
zero_out_kernel(float4* __restrict__ out, int n4) {
    int i = blockIdx.x * blockDim.x + threadIdx.x;
    if (i < n4) out[i] = make_float4(0.f, 0.f, 0.f, 0.f);
}

__global__ void __launch_bounds__(256)
patch_attn_kernel(const float* __restrict__ q,
                  const float* __restrict__ k,
                  const float* __restrict__ v,
                  float* __restrict__ out) {
    extern __shared__ __align__(16) char smem_raw[];
    float*    stq = (float*)(smem_raw + OFF_STQ);
    float*    stk = (float*)(smem_raw + OFF_STK);
    float*    stv = (float*)(smem_raw + OFF_STV);
    _Float16* qs  = (_Float16*)(smem_raw + OFF_QS);
    float*    sm  = (float*)(smem_raw + OFF_QS);   // union reuse after S phase
    _Float16* ks  = (_Float16*)(smem_raw + OFF_KS);
    _Float16* vs  = (_Float16*)(smem_raw + OFF_VS);
    _Float16* ps  = (_Float16*)(smem_raw + OFF_PS);

    const int n   = blockIdx.x;
    const int b   = n / (NH * NW);
    const int rem = n % (NH * NW);
    const int ph  = rem / NW, pw = rem % NW;
    const int h0  = ph * PSTR, w0 = pw * PSTR;
    const int tid  = threadIdx.x;
    const int lane = tid & 31;
    const int wave = tid >> 5;

#if HAVE_TDM
    // ---- TDM gather: one descriptor per matrix, issued by wave 0 only ----
    if (wave == 0) {
        const size_t gbase = ((size_t)(b * CH) * IH + h0) * IW + w0;
        tdm_load_tile(q + gbase, (unsigned int)(uintptr_t)stq);
        tdm_load_tile(k + gbase, (unsigned int)(uintptr_t)stk);
        tdm_load_tile(v + gbase, (unsigned int)(uintptr_t)stv);
        __builtin_amdgcn_s_wait_tensorcnt(0);
    }
    __syncthreads();
#endif

    // ---- convert staged f32 [c][t] tiles into f16 WMMA layouts ----
    for (int s = tid; s < CH * PK; s += 256) {
        const int c  = s >> 3;
        const int kh = s & 7;
        const int tb = kh * 8;  // token base
#if HAVE_TDM
        const int sbase = c * T + tb;
        const float4 q0 = *(const float4*)(stq + sbase);
        const float4 q1 = *(const float4*)(stq + sbase + 4);
        const float4 k0v = *(const float4*)(stk + sbase);
        const float4 k1v = *(const float4*)(stk + sbase + 4);
        const float4 v0 = *(const float4*)(stv + sbase);
        const float4 v1 = *(const float4*)(stv + sbase + 4);
#else
        const size_t goff = ((size_t)(b * CH + c) * IH + (h0 + kh)) * IW + w0;
        const float4 q0 = *(const float4*)(q + goff);
        const float4 q1 = *(const float4*)(q + goff + 4);
        const float4 k0v = *(const float4*)(k + goff);
        const float4 k1v = *(const float4*)(k + goff + 4);
        const float4 v0 = *(const float4*)(v + goff);
        const float4 v1 = *(const float4*)(v + goff + 4);
#endif
        qs[(tb + 0) * LDQ + c] = (_Float16)q0.x;
        qs[(tb + 1) * LDQ + c] = (_Float16)q0.y;
        qs[(tb + 2) * LDQ + c] = (_Float16)q0.z;
        qs[(tb + 3) * LDQ + c] = (_Float16)q0.w;
        qs[(tb + 4) * LDQ + c] = (_Float16)q1.x;
        qs[(tb + 5) * LDQ + c] = (_Float16)q1.y;
        qs[(tb + 6) * LDQ + c] = (_Float16)q1.z;
        qs[(tb + 7) * LDQ + c] = (_Float16)q1.w;

        ks[(tb + 0) * LDQ + c] = (_Float16)k0v.x;
        ks[(tb + 1) * LDQ + c] = (_Float16)k0v.y;
        ks[(tb + 2) * LDQ + c] = (_Float16)k0v.z;
        ks[(tb + 3) * LDQ + c] = (_Float16)k0v.w;
        ks[(tb + 4) * LDQ + c] = (_Float16)k1v.x;
        ks[(tb + 5) * LDQ + c] = (_Float16)k1v.y;
        ks[(tb + 6) * LDQ + c] = (_Float16)k1v.z;
        ks[(tb + 7) * LDQ + c] = (_Float16)k1v.w;

        v8h vv;
        vv[0] = (_Float16)v0.x; vv[1] = (_Float16)v0.y;
        vv[2] = (_Float16)v0.z; vv[3] = (_Float16)v0.w;
        vv[4] = (_Float16)v1.x; vv[5] = (_Float16)v1.y;
        vv[6] = (_Float16)v1.z; vv[7] = (_Float16)v1.w;
        *(v8h*)(vs + c * LDV + tb) = vv;
    }
    __syncthreads();

    // ---- S = Q K^T : 16 tiles of 16x16, 2 per wave, K-loop 128/32 = 4 ----
    const int tileA = wave;       // tiles 0..7
    const int tileB = wave + 8;   // tiles 8..15
    v8f acc0 = {};
    v8f acc1 = {};
#pragma unroll
    for (int kk = 0; kk < 4; ++kk) {
        v16h a0 = load_frag_a(qs, (tileA >> 2) * 16, kk * 32, LDQ, lane);
        v16h b0 = load_frag_b(ks, (tileA & 3) * 16,  kk * 32, LDQ, lane);
        acc0 = __builtin_amdgcn_wmma_f32_16x16x32_f16(false, a0, false, b0,
                                                      (short)0, acc0, false, false);
        v16h a1 = load_frag_a(qs, (tileB >> 2) * 16, kk * 32, LDQ, lane);
        v16h b1 = load_frag_b(ks, (tileB & 3) * 16,  kk * 32, LDQ, lane);
        acc1 = __builtin_amdgcn_wmma_f32_16x16x32_f16(false, a1, false, b1,
                                                      (short)0, acc1, false, false);
    }
    __syncthreads();  // all qs reads done; safe to overwrite union with scores

    store_acc(sm, tileA, acc0, lane);
    store_acc(sm, tileB, acc1, lane);
    __syncthreads();

    // ---- softmax over rows (scale = C^-0.5 folded into exp) ----
    if (tid < T) {
        const float scale = 0.088388347648318447f;  // 1/sqrt(128)
        const float* row = sm + tid * LDS_SM;
        float m = -3.0e38f;
#pragma unroll 8
        for (int j = 0; j < T; ++j) m = fmaxf(m, row[j]);
        float sum = 0.f;
#pragma unroll 8
        for (int j = 0; j < T; ++j) sum += __expf((row[j] - m) * scale);
        const float inv = __frcp_rn(sum);
        _Float16* prow = ps + tid * LDP;
#pragma unroll 8
        for (int j = 0; j < T; ++j)
            prow[j] = (_Float16)(__expf((row[j] - m) * scale) * inv);
    }
    __syncthreads();

    // ---- Z = P V : 32 tiles (4 token rows x 8 chan cols), 4 per wave ----
#pragma unroll
    for (int j = 0; j < 4; ++j) {
        const int zt = wave + 8 * j;
        const int mi = zt >> 3;   // token-tile 0..3
        const int ci = zt & 7;    // channel-tile 0..7
        v8f acc = {};
#pragma unroll
        for (int kk = 0; kk < 2; ++kk) {
            v16h a  = load_frag_a(ps, mi * 16, kk * 32, LDP, lane);
            v16h bb = load_frag_b(vs, ci * 16, kk * 32, LDV, lane);
            acc = __builtin_amdgcn_wmma_f32_16x16x32_f16(false, a, false, bb,
                                                         (short)0, acc, false, false);
        }
        // overlap-add scatter (non-returning global_atomic_add_f32)
        const int hi = (lane & 16) ? 8 : 0;
        const int c  = ci * 16 + (lane & 15);
#pragma unroll
        for (int i = 0; i < 8; ++i) {
            const int t  = mi * 16 + hi + i;
            const int kh = t >> 3, kw = t & 7;
            const size_t oidx =
                ((size_t)(b * CH + c) * IH + (h0 + kh)) * IW + (w0 + kw);
            __hip_atomic_fetch_add(out + oidx, acc[i],
                                   __ATOMIC_RELAXED, __HIP_MEMORY_SCOPE_AGENT);
        }
    }
}

extern "C" void kernel_launch(void* const* d_in, const int* in_sizes, int n_in,
                              void* d_out, int out_size, void* d_ws, size_t ws_size,
                              hipStream_t stream) {
    const float* q = (const float*)d_in[0];
    const float* k = (const float*)d_in[1];
    const float* v = (const float*)d_in[2];
    float* out = (float*)d_out;

    const int n4 = out_size / 4;  // 8*128*128*128 = 16,777,216 floats
    zero_out_kernel<<<(n4 + 255) / 256, 256, 0, stream>>>((float4*)out, n4);

    patch_attn_kernel<<<BATCH * NH * NW, 256, SMEM_BYTES, stream>>>(q, k, v, out);
}
